// LSTM_AE_74715251081632
// MI455X (gfx1250) — matrix operand order repaired and checked by reading, not compiled, not run on hardware
//
#include <hip/hip_runtime.h>
#include <hip/hip_bf16.h>
#include <stdint.h>

typedef __attribute__((ext_vector_type(16))) __bf16 v16bf;
typedef __attribute__((ext_vector_type(8)))  float  v8f;
typedef unsigned short us;

#define T_LEN 2048
#define BATCH 64
#define HID   128
#define G4    512    // 4*H
#define LP    136    // padded LDS row stride (272B -> bank-conflict-free frags)

__device__ __forceinline__ us f32_bf16(float f) {
  union { float f; unsigned int u; } x; x.f = f;
  unsigned int lsb = (x.u >> 16) & 1u;
  x.u += 0x7fffu + lsb;
  return (us)(x.u >> 16);
}

__device__ __forceinline__ float fast_sig(float x) {
  // 1/(1+e^-x): mul, v_exp, add, v_rcp
  return __builtin_amdgcn_rcpf(1.0f + __expf(-x));
}
__device__ __forceinline__ float fast_tanh(float x) {
  // 1 - 2/(e^{2x}+1): saturates cleanly at +-1
  return 1.0f - 2.0f * __builtin_amdgcn_rcpf(__expf(2.0f * x) + 1.0f);
}

// A-matrix fragment (16x32 bf16, MxK): lanes 0-15 hold K{0..7,16..23},
// lanes 16-31 hold K{8..15,24..31} of row M=lane&15 (ISA 7.12.2).
__device__ __forceinline__ v16bf load_afrag(const us* row, int k0, int lhi) {
  union { uint4 u[2]; v16bf v; } r;
  r.u[0] = *(const uint4*)(row + k0 + lhi * 8);
  r.u[1] = *(const uint4*)(row + k0 + 16 + lhi * 8);
  return r.v;
}
// B-matrix fragment (32x16 bf16, KxN): column N=lane&15; lanes 0-15 hold
// K{0..7,16..23}, lanes 16-31 hold K{8..15,24..31} (matches A fragment).
__device__ __forceinline__ v16bf load_bfrag(const us* col, int k0, int lhi) {
  union { uint4 u[2]; v16bf v; } r;
  r.u[0] = *(const uint4*)(col + k0 + lhi * 8);
  r.u[1] = *(const uint4*)(col + k0 + 16 + lhi * 8);
  return r.v;
}

// ---------------- conversion / prep kernels ----------------

__global__ void k_f32_to_bf16(const float* __restrict__ in, us* __restrict__ out, int n) {
  int i = blockIdx.x * blockDim.x + threadIdx.x;
  if (i < n) out[i] = f32_bf16(in[i]);
}

// x: [B][T][F] f32 -> xb: [(t*B+b)][F] bf16  (TB row order for the GEMMs)
__global__ void k_transpose_x(const float* __restrict__ x, us* __restrict__ xb) {
  int i = blockIdx.x * blockDim.x + threadIdx.x;   // over B*T*F
  int f = i & (HID - 1);
  int r = i >> 7;            // t*B + b
  int t = r >> 6;
  int b = r & 63;
  xb[i] = f32_bf16(x[((size_t)b * T_LEN + t) * HID + f]);
}

__global__ void k_bias_sum(const float* __restrict__ a, const float* __restrict__ b,
                           float* __restrict__ o, int n) {
  int i = blockIdx.x * blockDim.x + threadIdx.x;
  if (i < n) o[i] = a[i] + b[i];
}

// ---------------- big parallel GEMM: C[M][N] = A[M][128] @ W[N][128]^T + bias ----------------
// 256 threads (8 waves), tile 128(M) x 64(N), K=128 (4 wmma k-blocks), padded LDS

__global__ __launch_bounds__(256) void k_gemm_bias(
    const us* __restrict__ A, const us* __restrict__ W,
    const float* __restrict__ bias, float* __restrict__ C,
    int N, int remap_out) {
  __shared__ us a_s[128 * LP];   // ~34 KB
  __shared__ us w_s[64 * LP];    // ~17 KB
  const int tid = threadIdx.x;
  const int mblk = blockIdx.x, nblk = blockIdx.y;
  {
    const uint4* ga = (const uint4*)(A + (size_t)mblk * 128 * 128);
#pragma unroll
    for (int k = 0; k < 8; ++k) {
      int idx = tid + 256 * k;          // [128 rows][16 uint4]
      int row = idx >> 4, c = idx & 15;
      *(uint4*)(a_s + row * LP + c * 8) = ga[idx];
    }
    const uint4* gw = (const uint4*)(W + (size_t)nblk * 64 * 128);
#pragma unroll
    for (int k = 0; k < 4; ++k) {
      int idx = tid + 256 * k;          // [64 rows][16 uint4]
      int row = idx >> 4, c = idx & 15;
      *(uint4*)(w_s + row * LP + c * 8) = gw[idx];
    }
  }
  __syncthreads();

  const int wv = tid >> 5, lane = tid & 31;
  const int l15 = lane & 15, lhi = lane >> 4;
  const int wcol = wv & 3, wrow = wv >> 2;
  const int n0 = wcol * 16;
  const float bv = bias[nblk * 64 + n0 + l15];

  v8f acc[4];
#pragma unroll
  for (int m = 0; m < 4; ++m)
#pragma unroll
    for (int r = 0; r < 8; ++r) acc[m][r] = bv;

#pragma unroll
  for (int kb = 0; kb < 4; ++kb) {
    v16bf bf = load_bfrag(&w_s[(n0 + l15) * LP], kb * 32, lhi);
#pragma unroll
    for (int m = 0; m < 4; ++m) {
      v16bf af = load_afrag(&a_s[(wrow * 64 + m * 16 + l15) * LP], kb * 32, lhi);
      acc[m] = __builtin_amdgcn_wmma_f32_16x16x32_bf16(
          false, af, false, bf, (short)0, acc[m], false, false);
    }
  }

#pragma unroll
  for (int m = 0; m < 4; ++m) {
#pragma unroll
    for (int r = 0; r < 8; ++r) {
      int grow = mblk * 128 + wrow * 64 + m * 16 + r + lhi * 8;
      int gcol = nblk * 64 + n0 + l15;
      size_t off;
      if (remap_out) {  // TB row -> [B][T][N]
        int t = grow >> 6, b = grow & 63;
        off = ((size_t)b * T_LEN + t) * N + gcol;
      } else {
        off = (size_t)grow * N + gcol;
      }
      C[off] = acc[m][r];
    }
  }
}

// ---------------- sequential LSTM scan ----------------
// 1 workgroup, 32 waves. Wave w owns batch-row tile mt=w>>3 and hidden-col tile
// jt=w&7, computing all 4 gates for that (row,col) tile -> cell update entirely
// in registers. h double-buffered in LDS -> ONE barrier per step.
// LDS: Whh padded (136KB) + 2x h (17KB each) = 170KB

__global__ __launch_bounds__(1024) void k_lstm_scan(
    const float* __restrict__ pre,   // [T][B][4H] pre-gates (incl bias) or nullptr
    const us* __restrict__ whh,      // [512][128] bf16
    const float* __restrict__ bias,  // [512] (used when pre==nullptr)
    const float* __restrict__ h0,    // [B][H] or nullptr (zeros)
    const float* __restrict__ c0,    // [B][H] or nullptr (zeros)
    us* __restrict__ y,              // [T][B][H] bf16
    float* __restrict__ hT, float* __restrict__ cT) {
  extern __shared__ char smem[];
  us* whh_s = (us*)smem;                                   // 512*LP us
  us* hbuf0 = (us*)(smem + 512 * LP * 2);                  // 64*LP us
  us* hbuf1 = (us*)(smem + 512 * LP * 2 + 64 * LP * 2);    // 64*LP us
  const int tid = threadIdx.x;

  {  // stage Whh into padded LDS rows: 512 rows x 16 uint4
    const uint4* g = (const uint4*)whh;
#pragma unroll
    for (int k = 0; k < 8; ++k) {
      int idx = tid + 1024 * k;
      int row = idx >> 4, c = idx & 15;
      *(uint4*)(whh_s + row * LP + c * 8) = g[idx];
    }
  }

  const int wv = tid >> 5, lane = tid & 31;
  const int l15 = lane & 15, lhi = lane >> 4;
  const int mt = wv >> 3;             // batch-row tile 0..3
  const int jt = wv & 7;              // hidden-col tile 0..7
  const int col = jt * 16 + l15;      // hidden column 0..127

  float creg[8];
#pragma unroll
  for (int e = 0; e < 8; ++e) {       // init h buffer 0
    int idx = tid * 8 + e;
    int row = idx >> 7, cc = idx & 127;
    hbuf0[row * LP + cc] = h0 ? f32_bf16(h0[idx]) : (us)0;
  }
#pragma unroll
  for (int r = 0; r < 8; ++r) {       // c in registers, C-matrix layout
    int row = mt * 16 + r + lhi * 8;
    creg[r] = c0 ? c0[row * HID + col] : 0.0f;
  }
  const float bvi = bias[col], bvf = bias[128 + col],
              bvg = bias[256 + col], bvo = bias[384 + col];
  __syncthreads();

  us* hr = hbuf0;
  us* hw = hbuf1;
  for (int t = 0; t < T_LEN; ++t) {
    v8f gi, gf, gg, go;
    if (pre) {
      const float* pt = pre + (size_t)t * (BATCH * G4);
      if (t + 1 < T_LEN)
        __builtin_prefetch(pt + BATCH * G4 + tid * 32, 0, 0);
#pragma unroll
      for (int r = 0; r < 8; ++r) {
        int row = mt * 16 + r + lhi * 8;
        const float* pr = pt + (size_t)row * G4 + col;
        gi[r] = pr[0];
        gf[r] = pr[128];
        gg[r] = pr[256];
        go[r] = pr[384];
      }
    } else {
#pragma unroll
      for (int r = 0; r < 8; ++r) {
        gi[r] = bvi; gf[r] = bvf; gg[r] = bvg; go[r] = bvo;
      }
    }

#pragma unroll
    for (int kb = 0; kb < 4; ++kb) {
      v16bf a = load_afrag(&hr[(mt * 16 + l15) * LP], kb * 32, lhi);
      v16bf bi = load_bfrag(&whh_s[(0 * HID + col) * LP], kb * 32, lhi);
      v16bf bb = load_bfrag(&whh_s[(1 * HID + col) * LP], kb * 32, lhi);
      v16bf bg = load_bfrag(&whh_s[(2 * HID + col) * LP], kb * 32, lhi);
      v16bf bo = load_bfrag(&whh_s[(3 * HID + col) * LP], kb * 32, lhi);
      gi = __builtin_amdgcn_wmma_f32_16x16x32_bf16(false, a, false, bi, (short)0, gi, false, false);
      gf = __builtin_amdgcn_wmma_f32_16x16x32_bf16(false, a, false, bb, (short)0, gf, false, false);
      gg = __builtin_amdgcn_wmma_f32_16x16x32_bf16(false, a, false, bg, (short)0, gg, false, false);
      go = __builtin_amdgcn_wmma_f32_16x16x32_bf16(false, a, false, bo, (short)0, go, false, false);
    }

    // cell update entirely in registers (i,f,g,o share lane/VGPR slot with c)
    us* yt = y + (size_t)t * (BATCH * HID);
#pragma unroll
    for (int r = 0; r < 8; ++r) {
      int row = mt * 16 + r + lhi * 8;
      float iv = fast_sig(gi[r]);
      float fv = fast_sig(gf[r]);
      float gv = fast_tanh(gg[r]);
      float ov = fast_sig(go[r]);
      creg[r] = fv * creg[r] + iv * gv;
      float hv = ov * fast_tanh(creg[r]);
      us hb = f32_bf16(hv);
      hw[row * LP + col] = hb;            // write ping-pong buffer
      yt[row * HID + col] = hb;           // 16 lanes -> 32B contiguous
      if (t == T_LEN - 1) {
        hT[row * HID + col] = hv;
        cT[row * HID + col] = creg[r];
      }
    }
    __syncthreads();   // hw complete before it becomes hr
    us* tmp = hr; hr = hw; hw = tmp;
  }
}

// ---------------- host orchestration ----------------

extern "C" void kernel_launch(void* const* d_in, const int* in_sizes, int n_in,
                              void* d_out, int out_size, void* d_ws, size_t ws_size,
                              hipStream_t stream) {
  const float* x     = (const float*)d_in[0];
  const float* eWih0 = (const float*)d_in[1];
  const float* eWhh0 = (const float*)d_in[2];
  const float* ebih0 = (const float*)d_in[3];
  const float* ebhh0 = (const float*)d_in[4];
  const float* eWih1 = (const float*)d_in[5];
  const float* eWhh1 = (const float*)d_in[6];
  const float* ebih1 = (const float*)d_in[7];
  const float* ebhh1 = (const float*)d_in[8];
  // dec_Wih0 (d_in[9]) unused: decoder layer 0 input is all zeros
  const float* dWhh0 = (const float*)d_in[10];
  const float* dbih0 = (const float*)d_in[11];
  const float* dbhh0 = (const float*)d_in[12];
  const float* dWih1 = (const float*)d_in[13];
  const float* dWhh1 = (const float*)d_in[14];
  const float* dbih1 = (const float*)d_in[15];
  const float* dbhh1 = (const float*)d_in[16];
  const float* outW  = (const float*)d_in[17];
  const float* outB  = (const float*)d_in[18];

  char* ws = (char*)d_ws;
  size_t off = 0;
  auto alloc = [&](size_t bytes) -> void* {
    off = (off + 255) & ~(size_t)255;
    void* p = ws + off;
    off += bytes;
    return p;
  };

  float* pre  = (float*)alloc((size_t)T_LEN * BATCH * G4 * 4);   // 256 MB, reused
  us* act0    = (us*)alloc((size_t)T_LEN * BATCH * HID * 2);     // 32 MB
  us* act1    = (us*)alloc((size_t)T_LEN * BATCH * HID * 2);     // 32 MB (x_bf16 first)
  us* wih0b   = (us*)alloc(G4 * HID * 2);
  us* wih1b   = (us*)alloc(G4 * HID * 2);
  us* wih3b   = (us*)alloc(G4 * HID * 2);
  us* whh0b   = (us*)alloc(G4 * HID * 2);
  us* whh1b   = (us*)alloc(G4 * HID * 2);
  us* whh2b   = (us*)alloc(G4 * HID * 2);
  us* whh3b   = (us*)alloc(G4 * HID * 2);
  us* outwb   = (us*)alloc(HID * HID * 2);
  float* bs0  = (float*)alloc(G4 * 4);
  float* bs1  = (float*)alloc(G4 * 4);
  float* bs2  = (float*)alloc(G4 * 4);
  float* bs3  = (float*)alloc(G4 * 4);
  float* hT0  = (float*)alloc(BATCH * HID * 4);
  float* cT0  = (float*)alloc(BATCH * HID * 4);
  float* hT1  = (float*)alloc(BATCH * HID * 4);
  float* cT1  = (float*)alloc(BATCH * HID * 4);
  float* hTs  = (float*)alloc(BATCH * HID * 4);
  float* cTs  = (float*)alloc(BATCH * HID * 4);

  const int WN = G4 * HID;
  k_f32_to_bf16<<<(WN + 255) / 256, 256, 0, stream>>>(eWih0, wih0b, WN);
  k_f32_to_bf16<<<(WN + 255) / 256, 256, 0, stream>>>(eWih1, wih1b, WN);
  k_f32_to_bf16<<<(WN + 255) / 256, 256, 0, stream>>>(dWih1, wih3b, WN);
  k_f32_to_bf16<<<(WN + 255) / 256, 256, 0, stream>>>(eWhh0, whh0b, WN);
  k_f32_to_bf16<<<(WN + 255) / 256, 256, 0, stream>>>(eWhh1, whh1b, WN);
  k_f32_to_bf16<<<(WN + 255) / 256, 256, 0, stream>>>(dWhh0, whh2b, WN);
  k_f32_to_bf16<<<(WN + 255) / 256, 256, 0, stream>>>(dWhh1, whh3b, WN);
  k_f32_to_bf16<<<(HID * HID + 255) / 256, 256, 0, stream>>>(outW, outwb, HID * HID);
  k_bias_sum<<<2, 256, 0, stream>>>(ebih0, ebhh0, bs0, G4);
  k_bias_sum<<<2, 256, 0, stream>>>(ebih1, ebhh1, bs1, G4);
  k_bias_sum<<<2, 256, 0, stream>>>(dbih0, dbhh0, bs2, G4);
  k_bias_sum<<<2, 256, 0, stream>>>(dbih1, dbhh1, bs3, G4);
  k_transpose_x<<<(BATCH * T_LEN * HID) / 256, 256, 0, stream>>>(x, act1);

  const dim3 g512(T_LEN * BATCH / 128, G4 / 64);   // 1024 x 8
  const dim3 g128(T_LEN * BATCH / 128, HID / 64);  // 1024 x 2
  const size_t scan_lds = (512 * LP + 2 * 64 * LP) * 2;  // 170 KB

  // encoder layer 0
  k_gemm_bias<<<g512, 256, 0, stream>>>(act1, wih0b, bs0, pre, G4, 0);
  k_lstm_scan<<<1, 1024, scan_lds, stream>>>(pre, whh0b, bs0, nullptr, nullptr, act0, hT0, cT0);
  // encoder layer 1 (y discarded; only final h,c matter)
  k_gemm_bias<<<g512, 256, 0, stream>>>(act0, wih1b, bs1, pre, G4, 0);
  k_lstm_scan<<<1, 1024, scan_lds, stream>>>(pre, whh1b, bs1, nullptr, nullptr, act1, hT1, cT1);
  // decoder layer 0: zero input -> recurrent + bias only
  k_lstm_scan<<<1, 1024, scan_lds, stream>>>(nullptr, whh2b, bs2, hT0, cT0, act0, hTs, cTs);
  // decoder layer 1
  k_gemm_bias<<<g512, 256, 0, stream>>>(act0, wih3b, bs3, pre, G4, 0);
  k_lstm_scan<<<1, 1024, scan_lds, stream>>>(pre, whh3b, bs3, hT1, cT1, act1, hTs, cTs);
  // output projection, remapped to [B][T][F]
  k_gemm_bias<<<g128, 256, 0, stream>>>(act1, outwb, outB, (float*)d_out, HID, 1);
}
